// CurvatureEncodingLayer_38062000177651
// MI455X (gfx1250) — matrix operand order repaired and checked by reading, not compiled, not run on hardware
//
#include <hip/hip_runtime.h>
#include <math.h>

// CDNA5 / gfx1250: wave32, WMMA (not MFMA).
typedef __attribute__((ext_vector_type(16))) _Float16 v16h;
typedef __attribute__((ext_vector_type(8)))  float    v8f;
typedef __attribute__((ext_vector_type(4)))  int      v4i;   // native vector for NT b128 loads

#define DC       16
#define HIDDEN   32
#define PI_F     3.14159265358979323846f
#define LN_EPS_F 1e-5f

// ---------------------------------------------------------------------------
// Kernel 0: zero the workspace accumulators (deg[n], sum[n]) with b128 stores.
// Harness poisons d_ws with 0xAA and does not re-poison between replays.
// ---------------------------------------------------------------------------
__global__ void curv_zero_ws(float4* __restrict__ ws, int n4) {
  int i = blockIdx.x * blockDim.x + threadIdx.x;
  if (i < n4) ws[i] = make_float4(0.f, 0.f, 0.f, 0.f);
}

// ---------------------------------------------------------------------------
// Kernel 1: symmetric edge scatter.  32M edges -> 128M f32 atomics into 8MB
// of accumulators (L2-resident on the 192MB L2).  The 256MB edge-index stream
// is read-once, so load it non-temporally (TH=NT) to avoid evicting the hot
// atomic lines; the 4MB orc array is reused 64M times, so its gathers stay RT.
// edge_index is [2, E] flat: src = ei[0:E], dst = ei[E:2E].
// ---------------------------------------------------------------------------
__device__ __forceinline__ void curv_edge_accum(int s, int d,
                                                const float* __restrict__ orc,
                                                float* __restrict__ deg,
                                                float* __restrict__ sum) {
  atomicAdd(&deg[s], 1.0f);
  atomicAdd(&deg[d], 1.0f);
  atomicAdd(&sum[s], orc[d]);
  atomicAdd(&sum[d], orc[s]);
}

__global__ void curv_edge_scatter(const int* __restrict__ ei,
                                  const float* __restrict__ orc,
                                  float* __restrict__ deg,
                                  float* __restrict__ sum,
                                  int n_edges) {
  int e0 = (blockIdx.x * blockDim.x + threadIdx.x) * 4;
  if (e0 >= n_edges) return;
  if (e0 + 3 < n_edges) {
    // four edges per thread: NT b128 loads of the src and dst streams
    v4i s4 = __builtin_nontemporal_load((const v4i*)(ei + e0));
    v4i d4 = __builtin_nontemporal_load((const v4i*)(ei + n_edges + e0));
    curv_edge_accum(s4.x, d4.x, orc, deg, sum);
    curv_edge_accum(s4.y, d4.y, orc, deg, sum);
    curv_edge_accum(s4.z, d4.z, orc, deg, sum);
    curv_edge_accum(s4.w, d4.w, orc, deg, sum);
  } else {
    for (int e = e0; e < n_edges; ++e)
      curv_edge_accum(ei[e], ei[n_edges + e], orc, deg, sum);
  }
}

// ---------------------------------------------------------------------------
// Kernel 2: fused node pipeline.  One wave owns a 16-node tile:
//   encode -> Phi[16x16] (LDS) -> WMMA layer1 (two N-tiles) -> relu+bias (LDS)
//   -> WMMA layer2 -> LayerNorm via shfl_xor(16) -> +Phi residual -> NT store.
// 128 threads = 4 waves = 64 nodes per block.
// ---------------------------------------------------------------------------
__global__ void __launch_bounds__(128)
curv_node_mlp(const float* __restrict__ orc,
              const float* __restrict__ W1, const float* __restrict__ b1,
              const float* __restrict__ W2, const float* __restrict__ b2,
              const float* __restrict__ gamma, const float* __restrict__ beta,
              const float* __restrict__ deg, const float* __restrict__ sum,
              float* __restrict__ out, int n_nodes) {
  __shared__ float phi[4][16 * DC];       // Phi_base tile per wave
  __shared__ float hbuf[4][16 * HIDDEN];  // hidden activations per wave

  const int wave = threadIdx.x >> 5;
  const int lane = threadIdx.x & 31;
  const int m    = lane & 15;   // node row within tile / B column
  const int half = lane >> 4;   // lane-half selects K/encoding half
  const int tile_base = (blockIdx.x * 4 + wave) * 16;
  const int node  = tile_base + m;
  const int nodec = node < n_nodes ? node : (n_nodes - 1);  // clamp, keep EXEC full

  // ---- harmonic encoding: half 0 encodes orc, half 1 encodes nb_mean ------
  float x;
  if (half == 0) {
    x = orc[nodec];
  } else {
    float d = deg[nodec];
    x = (d > 0.0f) ? (sum[nodec] / d) : 0.0f;
  }
  float nrm = (x + 1.0f) * (1.0f / (2.0f + 1e-8f));
  nrm = fminf(fmaxf(nrm, 0.0f), 1.0f);
#pragma unroll
  for (int k = 1; k <= 4; ++k) {
    float ang = nrm * (PI_F * (float)k);
    phi[wave][m * DC + half * 8 + 2 * (k - 1)]     = sinf(ang);
    phi[wave][m * DC + half * 8 + 2 * (k - 1) + 1] = cosf(ang);
  }
  __syncthreads();

  // ---- pack A1 (16x32 f16, K = Phi features 0..15, 16..31 zero-padded) ----
  // ISA 16-bit A layout: lane m = lane&15; VGPR j<4 holds K = half*8+2j,2j+1;
  // VGPR j>=4 holds K = 16 + half*8 + ... (all zero here).
  v16h a1;
#pragma unroll
  for (int j = 0; j < 4; ++j) {
    int kb = half * 8 + 2 * j;
    a1[2 * j]     = (_Float16)phi[wave][m * DC + kb];
    a1[2 * j + 1] = (_Float16)phi[wave][m * DC + kb + 1];
  }
#pragma unroll
  for (int j = 8; j < 16; ++j) a1[j] = (_Float16)0.0f;

  // ---- pack B = W1^T, split into two N=16 tiles (HIDDEN=32) ---------------
  // B layout: column n = lane&15; slot i<8 holds K = 8*half + i, slot i>=8 holds K = 16 + 8*half + (i-8).
  v16h bw0, bw1;
#pragma unroll
  for (int i = 0; i < 16; ++i) {
    int K = (i < 8) ? (half * 8 + i) : (16 + half * 8 + (i - 8));
    float w0 = (K < DC) ? W1[m * DC + K] : 0.0f;          // rows h = 0..15
    float w1 = (K < DC) ? W1[(m + 16) * DC + K] : 0.0f;   // rows h = 16..31
    bw0[i] = (_Float16)w0;
    bw1[i] = (_Float16)w1;
  }

  v8f zacc = {};
  v8f acc0 = __builtin_amdgcn_wmma_f32_16x16x32_f16(false, a1, false, bw0,
                                                    (short)0, zacc, false, false);
  v8f acc1 = __builtin_amdgcn_wmma_f32_16x16x32_f16(false, a1, false, bw1,
                                                    (short)0, zacc, false, false);

  // ---- bias + relu, scatter C-layout rows into hbuf -----------------------
  // C/D layout: VGPR r -> row (r + 8*half), col = lane&15.
  float biasA = b1[m], biasB = b1[16 + m];
#pragma unroll
  for (int r = 0; r < 8; ++r) {
    int row = r + 8 * half;
    hbuf[wave][row * HIDDEN + m]      = fmaxf(acc0[r] + biasA, 0.0f);
    hbuf[wave][row * HIDDEN + 16 + m] = fmaxf(acc1[r] + biasB, 0.0f);
  }
  __syncthreads();

  // ---- pack A2 (16x32 f16 from hidden acts) and B2 = W2^T -----------------
  v16h a2, bw2;
#pragma unroll
  for (int j = 0; j < 8; ++j) {
    int kb = (j < 4) ? (half * 8 + 2 * j) : (16 + half * 8 + 2 * (j - 4));
    a2[2 * j]     = (_Float16)hbuf[wave][m * HIDDEN + kb];
    a2[2 * j + 1] = (_Float16)hbuf[wave][m * HIDDEN + kb + 1];
  }
#pragma unroll
  for (int i = 0; i < 16; ++i) {
    int K = (i < 8) ? (half * 8 + i) : (16 + half * 8 + (i - 8));
    bw2[i] = (_Float16)W2[m * HIDDEN + K];  // W2 is [DC, HIDDEN] row-major
  }
  v8f acc2 = __builtin_amdgcn_wmma_f32_16x16x32_f16(false, a2, false, bw2,
                                                    (short)0, zacc, false, false);

  // ---- bias2 + LayerNorm(16) via shfl_xor within each 16-lane half --------
  float g = gamma[m], bb = beta[m], bias2 = b2[m];
#pragma unroll
  for (int r = 0; r < 8; ++r) {
    float y  = acc2[r] + bias2;
    float s1 = y, s2 = y * y;
#pragma unroll
    for (int mask = 1; mask <= 8; mask <<= 1) {
      s1 += __shfl_xor(s1, mask, 32);
      s2 += __shfl_xor(s2, mask, 32);
    }
    float mu  = s1 * (1.0f / 16.0f);
    float var = s2 * (1.0f / 16.0f) - mu * mu;
    float rs  = rsqrtf(var + LN_EPS_F);
    int row   = r + 8 * half;
    float o   = (y - mu) * rs * g + bb + phi[wave][row * DC + m];
    int nr    = tile_base + row;
    if (nr < n_nodes)  // write-once output: NT store, don't pollute L2
      __builtin_nontemporal_store(o, &out[nr * DC + m]);
  }
}

// ---------------------------------------------------------------------------
extern "C" void kernel_launch(void* const* d_in, const int* in_sizes, int n_in,
                              void* d_out, int out_size, void* d_ws, size_t ws_size,
                              hipStream_t stream) {
  const float* orc   = (const float*)d_in[0];
  const int*   ei    = (const int*)d_in[1];
  const float* W1    = (const float*)d_in[2];
  const float* b1    = (const float*)d_in[3];
  const float* W2    = (const float*)d_in[4];
  const float* b2    = (const float*)d_in[5];
  const float* gamma = (const float*)d_in[6];
  const float* beta  = (const float*)d_in[7];
  float*       out   = (float*)d_out;

  const int n_nodes = in_sizes[0];
  const int n_edges = in_sizes[1] / 2;

  float* deg = (float*)d_ws;       // [n_nodes]
  float* sum = deg + n_nodes;      // [n_nodes]

  {
    int n4     = (2 * n_nodes + 3) / 4;  // 2*n_nodes floats, b128 stores
    int blocks = (n4 + 255) / 256;
    curv_zero_ws<<<blocks, 256, 0, stream>>>((float4*)d_ws, n4);
  }
  {
    int quads  = (n_edges + 3) / 4;
    int blocks = (quads + 255) / 256;
    curv_edge_scatter<<<blocks, 256, 0, stream>>>(ei, orc, deg, sum, n_edges);
  }
  {
    int blocks = (n_nodes + 63) / 64;  // 64 nodes per 128-thread block
    curv_node_mlp<<<blocks, 128, 0, stream>>>(orc, W1, b1, W2, b2, gamma, beta,
                                              deg, sum, out, n_nodes);
  }
}
